// QNet_mimo_graph_29755533427211
// MI455X (gfx1250) — matrix operand order repaired and checked by reading, not compiled, not run on hardware
//
#include <hip/hip_runtime.h>

typedef __attribute__((ext_vector_type(16))) __bf16 v16bf;
typedef __attribute__((ext_vector_type(8)))  float  v8f;

#define KDIM 128
#define PDIM 64
#define MUSTRIDE (PDIM + 2)   // stride 66: conflict-free C-layout stores, even for b64 merges

__global__ __launch_bounds__(256)
void qnet_mimo_kernel(const float* __restrict__ x,
                      const float* __restrict__ w,
                      const float* __restrict__ W1,
                      const float* __restrict__ W2,
                      const float* __restrict__ W3,
                      const float* __restrict__ W4,
                      const float* __restrict__ W5,
                      const float* __restrict__ W6,
                      const float* __restrict__ W7,
                      float* __restrict__ out)
{
    __shared__ float mu[KDIM][MUSTRIDE];
    __shared__ float wbuf[KDIM], xbuf[KDIM];
    __shared__ float Apos[PDIM], Aneg[PDIM], v7s[PDIM], S3s[PDIM];
    __shared__ float part[8][PDIM];        // per-wave column partials
    __shared__ float colsum[PDIM], pooled[PDIM];
    __shared__ float WposS, WnegS, scalS;

    const int tid  = threadIdx.x;
    const int lane = tid & 31;
    const int wv   = tid >> 5;
    const int b    = blockIdx.x;
    const int half = lane >> 4;
    const int l15  = lane & 15;
    const int m0   = wv * 16;              // this wave's 16 rows of mu

    // ---- per-b inputs ----
    if (tid < KDIM) {
        wbuf[tid] = w[b * KDIM + tid];
        xbuf[tid] = x[b * KDIM + tid];
    }

    // ---- resident W2^T bf16 B-fragments: B[kq,n] = W2[n,kq] ----
    v16bf bfrag[4][2];
    #pragma unroll
    for (int nt = 0; nt < 4; ++nt) {
        const int n = nt * 16 + l15;
        #pragma unroll
        for (int c = 0; c < 2; ++c) {
            const int k0 = c * 32;
            #pragma unroll
            for (int v = 0; v < 8; ++v) {
                const int kb = k0 + 2 * v + 16 * half;
                bfrag[nt][c][2 * v]     = (__bf16)W2[n * PDIM + kb];
                bfrag[nt][c][2 * v + 1] = (__bf16)W2[n * PDIM + kb + 1];
            }
        }
    }

    __syncthreads();

    // ---- scalar collapses of the tmsg/W4 structure ----
    if (tid == 0) { float s = 0.f; for (int k = 0; k < KDIM; ++k) { float v = wbuf[k]; if (v > 0.f) s += v; } WposS = s; }
    if (tid == 1) { float s = 0.f; for (int k = 0; k < KDIM; ++k) { float v = wbuf[k]; if (v < 0.f) s += v; } WnegS = s; }
    if (tid < PDIM) {
        float ap = 0.f, an = 0.f, vv = 0.f;
        for (int q = 0; q < PDIM; ++q) {
            const float w4 = W4[q];
            const float w3 = W3[tid * PDIM + q];
            if (w4 > 0.f) ap += w4 * w3; else an += w4 * w3;
            vv += W5[PDIM + q] * W7[q * PDIM + tid];        // v7 = W5b @ W7
        }
        Apos[tid] = ap; Aneg[tid] = an; v7s[tid] = vv;
    }
    __syncthreads();
    if (tid < PDIM) {                                        // S3 = sum_t @ W3^T
        const float wp = WposS, wn = WnegS;
        float s3 = 0.f;
        for (int p = 0; p < PDIM; ++p) {
            const float w4 = W4[p];
            s3 += (w4 * (w4 > 0.f ? wp : wn)) * W3[tid * PDIM + p];
        }
        S3s[tid] = s3;
    }
    __syncthreads();

    // ---- c3 in registers (C-tile layout), computed ONCE ----
    float c3r[4][8], muv[4][8];
    float W1c[4], S3c[4], Apc[4], Anc[4];
    #pragma unroll
    for (int nt = 0; nt < 4; ++nt) {
        const int col = nt * 16 + l15;
        W1c[nt] = W1[col]; S3c[nt] = S3s[col];
        Apc[nt] = Apos[col]; Anc[nt] = Aneg[col];
    }
    float xr[8], wr[8];
    #pragma unroll
    for (int r = 0; r < 8; ++r) {
        const int row = m0 + r + 8 * half;
        xr[r] = xbuf[row]; wr[r] = wbuf[row];
    }
    #pragma unroll
    for (int nt = 0; nt < 4; ++nt)
        #pragma unroll
        for (int r = 0; r < 8; ++r) {
            const float a = (wr[r] > 0.f) ? Apc[nt] : Anc[nt];
            c3r[nt][r] = xr[r] * W1c[nt] + S3c[nt] - wr[r] * a;
            muv[nt][r] = fmaxf(c3r[nt][r], 0.f);             // iteration 1: mu = relu(c3)
        }

    // store this wave's rows + per-wave column partials (registers + shfl, no LDS sweep)
    auto store_and_partial = [&]() {
        #pragma unroll
        for (int nt = 0; nt < 4; ++nt) {
            float s = 0.f;
            #pragma unroll
            for (int r = 0; r < 8; ++r) {
                mu[m0 + r + 8 * half][nt * 16 + l15] = muv[nt][r];
                s += muv[nt][r];
            }
            s += __shfl_xor(s, 16, 32);                      // combine lane halves (16 rows)
            if (lane < 16) part[wv][nt * 16 + lane] = s;
        }
    };
    store_and_partial();
    __syncthreads();

    // ---- iterations 2..4: mu = relu(c3 + (colsum - mu) @ W2^T) ----
    for (int it = 0; it < 3; ++it) {
        if (tid < PDIM) {
            float cs = 0.f;
            #pragma unroll
            for (int wq = 0; wq < 8; ++wq) cs += part[wq][tid];
            colsum[tid] = cs;
        }
        __syncthreads();

        // A' = colsum - mu (own rows), bf16 A-fragments (16-bit A 16x32 layout)
        v16bf afr[2];
        #pragma unroll
        for (int c = 0; c < 2; ++c) {
            const int k0 = c * 32;
            const int row = m0 + l15;
            #pragma unroll
            for (int v = 0; v < 8; ++v) {
                const int kb = k0 + 2 * v + 16 * half;
                afr[c][2 * v]     = (__bf16)(colsum[kb]     - mu[row][kb]);
                afr[c][2 * v + 1] = (__bf16)(colsum[kb + 1] - mu[row][kb + 1]);
            }
        }

        v8f acc[4];
        #pragma unroll
        for (int nt = 0; nt < 4; ++nt) {
            v8f cz = {0.f, 0.f, 0.f, 0.f, 0.f, 0.f, 0.f, 0.f};
            cz = __builtin_amdgcn_wmma_f32_16x16x32_bf16(false, afr[0], false, bfrag[nt][0], (short)0, cz, false, false);
            cz = __builtin_amdgcn_wmma_f32_16x16x32_bf16(false, afr[1], false, bfrag[nt][1], (short)0, cz, false, false);
            acc[nt] = cz;
        }

        #pragma unroll
        for (int nt = 0; nt < 4; ++nt)
            #pragma unroll
            for (int r = 0; r < 8; ++r)
                muv[nt][r] = fmaxf(c3r[nt][r] + acc[nt][r], 0.f);

        store_and_partial();
        __syncthreads();
    }

    // ---- readout ----
    if (tid < PDIM) {                                        // m = final column sums (from parts)
        float m = 0.f;
        #pragma unroll
        for (int wq = 0; wq < 8; ++wq) m += part[wq][tid];
        colsum[tid] = m;
    }
    __syncthreads();
    if (tid < PDIM) {                                        // pooled = relu(m @ W6^T)
        float s = 0.f;
        for (int p = 0; p < PDIM; ++p) s += colsum[p] * W6[tid * PDIM + p];
        pooled[tid] = fmaxf(s, 0.f);
    }
    __syncthreads();
    if (tid == 0) {
        float s = 0.f;
        for (int o = 0; o < PDIM; ++o) s += pooled[o] * W5[o];
        scalS = s;
    }
    __syncthreads();
    if (tid < KDIM) {                                        // q = scal + mu[k,:] . v7
        float s = scalS;
        for (int p = 0; p < PDIM; ++p) s += mu[tid][p] * v7s[p];
        out[b * KDIM + tid] = s;
    }
}

extern "C" void kernel_launch(void* const* d_in, const int* in_sizes, int n_in,
                              void* d_out, int out_size, void* d_ws, size_t ws_size,
                              hipStream_t stream) {
    const float* x  = (const float*)d_in[0];
    const float* w  = (const float*)d_in[1];
    const float* W1 = (const float*)d_in[2];
    const float* W2 = (const float*)d_in[3];
    const float* W3 = (const float*)d_in[4];
    const float* W4 = (const float*)d_in[5];
    const float* W5 = (const float*)d_in[6];
    const float* W6 = (const float*)d_in[7];
    const float* W7 = (const float*)d_in[8];
    float* out = (float*)d_out;
    const int B = in_sizes[0] / KDIM;   // 4096
    qnet_mimo_kernel<<<B, 256, 0, stream>>>(x, w, W1, W2, W3, W4, W5, W6, W7, out);
    (void)n_in; (void)out_size; (void)d_ws; (void)ws_size;
}
